// RNNDecoder_21079699489073
// MI455X (gfx1250) — matrix operand order repaired and checked by reading, not compiled, not run on hardware
//
#include <hip/hip_runtime.h>
#include <hip/hip_bf16.h>
#include <math.h>

// ---------------------------------------------------------------------------
// CDNA5 / gfx1250 implementation: graph-GRU decoder.
// All big GEMMs use V_WMMA_F32_16X16X32_F16 (f16 operands, f32 accumulate).
// Operands are pre-packed into the wave32 WMMA fragment layout so that each
// lane loads its 16 halves as one contiguous 32B chunk (global/ds b128 pairs).
// ---------------------------------------------------------------------------

typedef __attribute__((ext_vector_type(16))) _Float16 v16h;
typedef __attribute__((ext_vector_type(8)))  float    v8f;

namespace {

constexpr int T_STEPS = 10;
constexpr int NNODES  = 10000;   // divisible by 16 -> 625 node tiles
constexpr int C_IN    = 4;
constexpr int H       = 128;
constexpr int E_EDGES = 160000;  // divisible by 32 -> 5000 edge tiles
constexpr int H2      = 2 * H;   // 256
constexpr int ET      = 32;      // edges per block (2 M-tiles)

// ---- fragment-native packed layouts (CDNA5 ISA 7.12.2, wave32) -------------
//
// A (16x32 f16 tile): lane = m + 16*hv, where lanes 0-15 hold K rem
// {0..7, 16..23} and lanes 16-31 hold {8..15, 24..31}. Each lane's 16 halves
// are stored contiguously -> one v16h per (ktile, lane).
__device__ __host__ inline int packA(int m, int k) {
  const int kt   = k >> 5;
  const int rem  = k & 31;
  const int hv   = (rem >> 3) & 1;
  const int slot = (rem < 16) ? (rem - 8 * hv) : (rem - 8 - 8 * hv);
  return ((kt * 32) + (m + 16 * hv)) * 16 + slot;
}

__device__ inline v8f wmma16(v16h a, v16h b, v8f c) {
  // 8 args: (neg_a, A, neg_b, B, c_mod, C, reuse_a, reuse_b)
  return __builtin_amdgcn_wmma_f32_16x16x32_f16(false, a, false, b, (short)0, c, false, false);
}

__device__ inline float sigm(float x) { return 1.0f / (1.0f + __expf(-x)); }

// ---------------------------------------------------------------------------
// Weight packing: f32 [batch][K][N] row-major -> f16 fragment-native B layout.
// B (32x16 f16 tile): lane = n + 16*hv with lanes 0-15 holding K rem
// {0..7, 16..23} and lanes 16-31 holding {8..15, 24..31} (same as packA).
// Packed index: (((nt*KT + kt)*32 + lane)*16 + slot), per-batch stride K*N.
// ---------------------------------------------------------------------------
__global__ void k_pack_b(const float* __restrict__ src, _Float16* __restrict__ dst,
                         int K, int N, int total) {
  const int idx = blockIdx.x * blockDim.x + threadIdx.x;
  if (idx >= total) return;
  const int per = K * N;
  const int b   = idx / per;
  const int r   = idx - b * per;
  const int k   = r / N;
  const int n   = r - k * N;
  const int kt = k >> 5, rem = k & 31, hv = (rem >> 3) & 1, slot = (rem < 16) ? (rem - 8 * hv) : (rem - 8 - 8 * hv);
  const int nt = n >> 4, nl = n & 15;
  const int lane = 16 * hv + nl;
  const int KT = K >> 5;
  dst[(size_t)b * per + (((nt * KT + kt) * 32) + lane) * 16 + slot] = (_Float16)src[idx];
}

// Narrow-N weight (out3_w: 128x4) packed into one zero-padded N-tile.
__global__ void k_pack_b_pad(const float* __restrict__ src, _Float16* __restrict__ dst,
                             int K, int Nsrc) {
  const int idx = blockIdx.x * blockDim.x + threadIdx.x;
  if (idx >= K * 16) return;
  const int k = idx >> 4;
  const int n = idx & 15;
  const float v = (n < Nsrc) ? src[k * Nsrc + n] : 0.0f;
  const int kt = k >> 5, rem = k & 31, hv = (rem >> 3) & 1, slot = (rem < 16) ? (rem - 8 * hv) : (rem - 8 - 8 * hv);
  const int lane = 16 * hv + n;
  dst[((kt * 32) + lane) * 16 + slot] = (_Float16)v;
}

__global__ void k_zero_f32(float* __restrict__ p, int n) {
  int i = blockIdx.x * blockDim.x + threadIdx.x;
  if (i < n) p[i] = 0.0f;
}

__global__ void k_select_x(const float* __restrict__ inputs, const float* __restrict__ pred_prev,
                           float* __restrict__ x_cur, const int* __restrict__ burn, int t) {
  int i = blockIdx.x * blockDim.x + threadIdx.x;
  if (i < NNODES * C_IN) {
    const bool use_in = (t < burn[0]) || (t == 0);
    if (use_in) x_cur[i] = inputs[(size_t)t * NNODES * C_IN + i];
    else        x_cur[i] = pred_prev[i];
  }
}

// ---------------------------------------------------------------------------
// Edge-message kernel: 32 edges per block (two 16-row M-tiles), 4 waves.
// Each wave owns 2 N-tiles; each B fragment load feeds 2 WMMAs (both M-tiles).
// all_msgs[e] = sum_k tanh( tanh(pre[e] @ W1[k] + b1[k]) @ W2[k] + b2[k] )
// then atomic scatter-add into agg[col[e]].
// ---------------------------------------------------------------------------
__global__ __launch_bounds__(128) void k_edge(
    const float* __restrict__ hidden, const int* __restrict__ edge_index,
    const v16h* __restrict__ w1p, const float* __restrict__ b1,
    const v16h* __restrict__ w2p, const float* __restrict__ b2,
    float* __restrict__ agg)
{
  __shared__ v16h preAp[2][8 * 32];   // packed A tiles, K=256 (8 k-tiles)
  __shared__ v16h mAp[2][4 * 32];     // packed intermediate, K=128 (4 k-tiles)
  __shared__ int rowi[ET], coli[ET];

  const int tid  = threadIdx.x;
  const int lane = tid & 31;
  const int wid  = tid >> 5;
  const int tile = blockIdx.x;

  __builtin_prefetch(w1p, 0, 1);      // global_prefetch_b8 on the streamed weights
  __builtin_prefetch(w2p, 0, 1);

  if (tid < ET) {
    rowi[tid] = edge_index[tile * ET + tid];
    coli[tid] = edge_index[E_EDGES + tile * ET + tid];
  }
  __syncthreads();

  // gather endpoint hidden states -> packed f16 A tiles
  for (int idx = tid; idx < ET * H2; idx += 128) {
    const int m = idx >> 8;          // edge row 0..31
    const int j = idx & (H2 - 1);    // feature 0..255
    const float v = (j < H) ? hidden[(size_t)rowi[m] * H + j]
                            : hidden[(size_t)coli[m] * H + (j - H)];
    ((_Float16*)&preAp[m >> 4][0])[packA(m & 15, j)] = (_Float16)v;
  }
  __syncthreads();

  const int hv  = (lane >> 4) & 1;
  const int nlo = lane & 15;

  float msum[2][2][8];   // [mtile][ntile-local][row]
#pragma unroll
  for (int mt = 0; mt < 2; ++mt)
#pragma unroll
    for (int lt = 0; lt < 2; ++lt)
#pragma unroll
      for (int r = 0; r < 8; ++r) msum[mt][lt][r] = 0.0f;

  for (int k = 0; k < 3; ++k) {
    const v16h* W1 = w1p + (size_t)k * (H2 * H / 16);  // 2048 fragments/branch
    const v16h* W2 = w2p + (size_t)k * (H * H / 16);   // 1024 fragments/branch

    // ---- GEMM1: [32,256] x [256,128] ----
    v8f acc[2][2];
#pragma unroll
    for (int lt = 0; lt < 2; ++lt) {
      const int nt = wid * 2 + lt;
      v8f c0 = {}, c1 = {};
#pragma unroll
      for (int kk = 0; kk < 8; ++kk) {
        const v16h b  = W1[(nt * 8 + kk) * 32 + lane];
        c0 = wmma16(preAp[0][kk * 32 + lane], b, c0);
        c1 = wmma16(preAp[1][kk * 32 + lane], b, c1);
      }
      acc[0][lt] = c0; acc[1][lt] = c1;
    }

    __syncthreads();  // previous branch's mAp reads complete
#pragma unroll
    for (int lt = 0; lt < 2; ++lt) {
      const int ncol = (wid * 2 + lt) * 16;
      const float bias = b1[k * H + ncol + nlo];
#pragma unroll
      for (int mt = 0; mt < 2; ++mt)
#pragma unroll
        for (int r = 0; r < 8; ++r)
          ((_Float16*)&mAp[mt][0])[packA(r + 8 * hv, ncol + nlo)] =
              (_Float16)tanhf(acc[mt][lt][r] + bias);
    }
    __syncthreads();  // mAp ready

    // ---- GEMM2: [32,128] x [128,128] ----
#pragma unroll
    for (int lt = 0; lt < 2; ++lt) {
      const int nt = wid * 2 + lt;
      const int ncol = nt * 16;
      v8f c0 = {}, c1 = {};
#pragma unroll
      for (int kk = 0; kk < 4; ++kk) {
        const v16h b = W2[(nt * 4 + kk) * 32 + lane];
        c0 = wmma16(mAp[0][kk * 32 + lane], b, c0);
        c1 = wmma16(mAp[1][kk * 32 + lane], b, c1);
      }
      const float bias = b2[k * H + ncol + nlo];
#pragma unroll
      for (int r = 0; r < 8; ++r) {
        msum[0][lt][r] += tanhf(c0[r] + bias);
        msum[1][lt][r] += tanhf(c1[r] + bias);
      }
    }
  }

  // scatter-add into receiver nodes
#pragma unroll
  for (int mt = 0; mt < 2; ++mt)
#pragma unroll
    for (int lt = 0; lt < 2; ++lt) {
      const int ncol = (wid * 2 + lt) * 16;
#pragma unroll
      for (int r = 0; r < 8; ++r) {
        const int e = mt * 16 + r + 8 * hv;
        atomicAdd(&agg[(size_t)coli[e] * H + ncol + nlo], msum[mt][lt][r]);
      }
    }
}

// ---------------------------------------------------------------------------
// Node-update kernel: tile of 16 nodes per block, 4 waves.
// GRU gates (agg @ hidden_*_w via WMMA, x-projection scalar since C=4),
// then 3-layer output MLP via WMMA, pred = x + relu-chain.
// ---------------------------------------------------------------------------
__global__ __launch_bounds__(128) void k_node(
    const float* __restrict__ x_cur, const float* __restrict__ agg,
    float* __restrict__ hidden,
    const v16h* __restrict__ hrp, const v16h* __restrict__ hip,
    const v16h* __restrict__ hhp,
    const float* __restrict__ irw, const float* __restrict__ irb,
    const float* __restrict__ iiw, const float* __restrict__ iib,
    const float* __restrict__ inw, const float* __restrict__ inb,
    const v16h* __restrict__ o1p, const float* __restrict__ o1b,
    const v16h* __restrict__ o2p, const float* __restrict__ o2b,
    const v16h* __restrict__ o3p, const float* __restrict__ o3b,
    float* __restrict__ out_t, float* __restrict__ pred_prev)
{
  __shared__ v16h aggAp[4 * 32];
  __shared__ v16h hidAp[4 * 32];
  __shared__ v16h p1Ap[4 * 32];
  __shared__ v16h p2Ap[4 * 32];
  __shared__ float xs[16][C_IN];

  const int tid   = threadIdx.x;
  const int lane  = tid & 31;
  const int wid   = tid >> 5;
  const int node0 = blockIdx.x * 16;

  for (int idx = tid; idx < 16 * H; idx += 128) {
    const int i = idx >> 7, j = idx & (H - 1);
    ((_Float16*)aggAp)[packA(i, j)] = (_Float16)agg[(size_t)(node0 + i) * H + j];
  }
  if (tid < 16 * C_IN) xs[tid >> 2][tid & 3] = x_cur[(size_t)node0 * C_IN + tid];
  __syncthreads();

  const int hv  = (lane >> 4) & 1;
  const int nlo = lane & 15;

  // ---- GRU gate GEMMs: agg @ {hr, hi, hh} ----
  v8f cr[2], ci[2], ch[2];
#pragma unroll
  for (int lt = 0; lt < 2; ++lt) {
    const int nt = wid * 2 + lt;
    v8f ar = {}, ai = {}, ah = {};
#pragma unroll
    for (int kk = 0; kk < 4; ++kk) {
      const v16h a = aggAp[kk * 32 + lane];
      ar = wmma16(a, hrp[(nt * 4 + kk) * 32 + lane], ar);
      ai = wmma16(a, hip[(nt * 4 + kk) * 32 + lane], ai);
      ah = wmma16(a, hhp[(nt * 4 + kk) * 32 + lane], ah);
    }
    cr[lt] = ar; ci[lt] = ai; ch[lt] = ah;
  }

  // ---- element-wise GRU update (column-local per wave) ----
#pragma unroll
  for (int lt = 0; lt < 2; ++lt) {
    const int n = (wid * 2 + lt) * 16 + nlo;
#pragma unroll
    for (int r = 0; r < 8; ++r) {
      const int mr   = r + 8 * hv;
      const int node = node0 + mr;
      float xr = irb[n], xi = iib[n], xn = inb[n];
#pragma unroll
      for (int c = 0; c < C_IN; ++c) {
        const float xv = xs[mr][c];
        xr += xv * irw[c * H + n];
        xi += xv * iiw[c * H + n];
        xn += xv * inw[c * H + n];
      }
      const float rg   = sigm(xr + cr[lt][r]);
      const float ig   = sigm(xi + ci[lt][r]);
      const float hold = hidden[(size_t)node * H + n];
      const float ng   = tanhf(xn + rg * ch[lt][r]);
      const float hnew = (1.0f - ig) * ng + ig * hold;
      hidden[(size_t)node * H + n] = hnew;
      ((_Float16*)hidAp)[packA(mr, n)] = (_Float16)hnew;
    }
  }
  __syncthreads();

  // ---- output MLP layer 1: relu(hidden @ out1_w + b) ----
#pragma unroll
  for (int lt = 0; lt < 2; ++lt) {
    const int nt = wid * 2 + lt;
    const int ncol = nt * 16;
    v8f c = {};
#pragma unroll
    for (int kk = 0; kk < 4; ++kk)
      c = wmma16(hidAp[kk * 32 + lane], o1p[(nt * 4 + kk) * 32 + lane], c);
    const float bias = o1b[ncol + nlo];
#pragma unroll
    for (int r = 0; r < 8; ++r)
      ((_Float16*)p1Ap)[packA(r + 8 * hv, ncol + nlo)] = (_Float16)fmaxf(c[r] + bias, 0.0f);
  }
  __syncthreads();

  // ---- layer 2 ----
#pragma unroll
  for (int lt = 0; lt < 2; ++lt) {
    const int nt = wid * 2 + lt;
    const int ncol = nt * 16;
    v8f c = {};
#pragma unroll
    for (int kk = 0; kk < 4; ++kk)
      c = wmma16(p1Ap[kk * 32 + lane], o2p[(nt * 4 + kk) * 32 + lane], c);
    const float bias = o2b[ncol + nlo];
#pragma unroll
    for (int r = 0; r < 8; ++r)
      ((_Float16*)p2Ap)[packA(r + 8 * hv, ncol + nlo)] = (_Float16)fmaxf(c[r] + bias, 0.0f);
  }
  __syncthreads();

  // ---- layer 3 (N = C_IN = 4, zero-padded tile) + residual, wave 0 only ----
  if (wid == 0) {
    v8f c = {};
#pragma unroll
    for (int kk = 0; kk < 4; ++kk)
      c = wmma16(p2Ap[kk * 32 + lane], o3p[kk * 32 + lane], c);
    if (nlo < C_IN) {
#pragma unroll
      for (int r = 0; r < 8; ++r) {
        const int mr   = r + 8 * hv;
        const int node = node0 + mr;
        const float p  = fmaxf(c[r] + o3b[nlo], 0.0f);
        const float pr = xs[mr][nlo] + p;
        out_t[(size_t)node * C_IN + nlo]     = pr;
        pred_prev[(size_t)node * C_IN + nlo] = pr;
      }
    }
  }
}

} // anonymous namespace

// ---------------------------------------------------------------------------

extern "C" void kernel_launch(void* const* d_in, const int* in_sizes, int n_in,
                              void* d_out, int out_size, void* d_ws, size_t ws_size,
                              hipStream_t stream) {
  (void)in_sizes; (void)n_in; (void)out_size; (void)ws_size;

  const float* inputs     = (const float*)d_in[0];
  const float* msg_fc1_w  = (const float*)d_in[1];
  const float* msg_fc1_b  = (const float*)d_in[2];
  const float* msg_fc2_w  = (const float*)d_in[3];
  const float* msg_fc2_b  = (const float*)d_in[4];
  const float* hidden_r_w = (const float*)d_in[5];
  const float* hidden_i_w = (const float*)d_in[6];
  const float* hidden_h_w = (const float*)d_in[7];
  const float* input_r_w  = (const float*)d_in[8];
  const float* input_r_b  = (const float*)d_in[9];
  const float* input_i_w  = (const float*)d_in[10];
  const float* input_i_b  = (const float*)d_in[11];
  const float* input_n_w  = (const float*)d_in[12];
  const float* input_n_b  = (const float*)d_in[13];
  const float* out1_w     = (const float*)d_in[14];
  const float* out1_b     = (const float*)d_in[15];
  const float* out2_w     = (const float*)d_in[16];
  const float* out2_b     = (const float*)d_in[17];
  const float* out3_w     = (const float*)d_in[18];
  const float* out3_b     = (const float*)d_in[19];
  const int*   edge_index = (const int*)d_in[20];
  const int*   burn_in    = (const int*)d_in[21];
  float* out = (float*)d_out;

  // ---- workspace carve (all segments 32B-aligned by construction) ----
  float* hidden    = (float*)d_ws;                       // N*H
  float* agg       = hidden + (size_t)NNODES * H;        // N*H
  float* x_cur     = agg + (size_t)NNODES * H;           // N*C
  float* pred_prev = x_cur + (size_t)NNODES * C_IN;      // N*C
  _Float16* w1p = (_Float16*)(pred_prev + (size_t)NNODES * C_IN);
  _Float16* w2p = w1p + 3 * H2 * H;
  _Float16* hrp = w2p + 3 * H * H;
  _Float16* hip = hrp + H * H;
  _Float16* hhp = hip + H * H;
  _Float16* o1p = hhp + H * H;
  _Float16* o2p = o1p + H * H;
  _Float16* o3p = o2p + H * H;   // H*16 (padded)

  auto cdiv = [](int a, int b) { return (a + b - 1) / b; };

  // ---- pack weights f32 -> fragment-native f16 (re-done every call) ----
  {
    const int n1 = 3 * H2 * H, n2 = 3 * H * H, n3 = H * H;
    k_pack_b<<<cdiv(n1, 256), 256, 0, stream>>>(msg_fc1_w, w1p, H2, H, n1);
    k_pack_b<<<cdiv(n2, 256), 256, 0, stream>>>(msg_fc2_w, w2p, H, H, n2);
    k_pack_b<<<cdiv(n3, 256), 256, 0, stream>>>(hidden_r_w, hrp, H, H, n3);
    k_pack_b<<<cdiv(n3, 256), 256, 0, stream>>>(hidden_i_w, hip, H, H, n3);
    k_pack_b<<<cdiv(n3, 256), 256, 0, stream>>>(hidden_h_w, hhp, H, H, n3);
    k_pack_b<<<cdiv(n3, 256), 256, 0, stream>>>(out1_w, o1p, H, H, n3);
    k_pack_b<<<cdiv(n3, 256), 256, 0, stream>>>(out2_w, o2p, H, H, n3);
    k_pack_b_pad<<<cdiv(H * 16, 256), 256, 0, stream>>>(out3_w, o3p, H, C_IN);
  }

  // hidden0 = 0
  k_zero_f32<<<cdiv(NNODES * H, 256), 256, 0, stream>>>(hidden, NNODES * H);

  for (int t = 0; t < T_STEPS; ++t) {
    k_select_x<<<cdiv(NNODES * C_IN, 256), 256, 0, stream>>>(inputs, pred_prev, x_cur, burn_in, t);
    k_zero_f32<<<cdiv(NNODES * H, 256), 256, 0, stream>>>(agg, NNODES * H);

    k_edge<<<E_EDGES / ET, 128, 0, stream>>>(hidden, edge_index,
                                             (const v16h*)w1p, msg_fc1_b,
                                             (const v16h*)w2p, msg_fc2_b, agg);

    k_node<<<NNODES / 16, 128, 0, stream>>>(
        x_cur, agg, hidden,
        (const v16h*)hrp, (const v16h*)hip, (const v16h*)hhp,
        input_r_w, input_r_b, input_i_w, input_i_b, input_n_w, input_n_b,
        (const v16h*)o1p, out1_b, (const v16h*)o2p, out2_b, (const v16h*)o3p, out3_b,
        out + (size_t)t * NNODES * C_IN, pred_prev);
  }
}
